// ArSSR_79233556677019
// MI455X (gfx1250) — compile-verified, matrix operand order checked by reading, not run on hardware
//
#include <hip/hip_runtime.h>
#include <hip/hip_bf16.h>

// ---------------------------------------------------------------------------
// Types for CDNA5 WMMA (wave32): v_wmma_f32_16x16x32_bf16
// ---------------------------------------------------------------------------
typedef __attribute__((ext_vector_type(16))) __bf16          v16bf;
typedef __attribute__((ext_vector_type(8)))  float           v8f;
typedef __attribute__((ext_vector_type(8)))  unsigned short  ushort8;

union FragBF {
    ushort8        u[2];
    unsigned short s[16];
    v16bf          v;
};

__device__ __forceinline__ unsigned short f2bf(float f) {
    unsigned u = __float_as_uint(f);
    unsigned r = (u + 0x7FFFu + ((u >> 16) & 1u)) >> 16;   // round-nearest-even
    return (unsigned short)r;
}

// K index inside a 16x32 bf16 A/B fragment for element i of a lane.
__device__ __forceinline__ int kmap(int i, int half) {
    return (i & 7) + ((i >> 3) << 4) + 8 * half;
}

// Permuted-space (c,z1,y1,x1) -> flat index in the ORIGINAL tensor layout.
// Identical formula for the input gather and the output scatter (the output
// permutation is the inverse of the input view-permutation).
__device__ __forceinline__ size_t perm_index(int br, int c, int z1, int y1, int x1) {
    switch (br) {
        case 0:  return ((size_t)(c * 48 + z1) * 66 + y1) * 38 + x1;  // axi (48,66,38)
        case 1:  return ((size_t)(c * 38 + z1) * 66 + x1) * 38 + y1;  // cor (38,66,38), view (0,1,2,4,3)
        default: return ((size_t)(c * 48 + x1) * 78 + y1) * 38 + z1;  // sag (48,78,38), view (0,1,4,3,2)
    }
}

// ---------------------------------------------------------------------------
// Fragment loaders
// ---------------------------------------------------------------------------
// A-operand from a row-major (rows x DP) bf16 buffer: two aligned B128 loads.
template <int DP>
__device__ __forceinline__ v16bf load_frag_rowmajor(const unsigned short* base,
                                                    int kk, int lane) {
    int row = lane & 15, half = lane >> 4;
    const unsigned short* p = base + (size_t)row * DP + kk + 8 * half;
    FragBF f;
    f.u[0] = *reinterpret_cast<const ushort8*>(p);
    f.u[1] = *reinterpret_cast<const ushort8*>(p + 16);
    return f.v;
}

// B-operand from the pre-packed lane-major layout: 32 contiguous bytes/lane.
template <int DP>
__device__ __forceinline__ v16bf load_frag_packed(const unsigned short* Wp,
                                                  int kt, int nt, int lane) {
    const unsigned short* p = Wp + ((size_t)(kt * (DP / 16) + nt) * 32 + lane) * 16;
    FragBF f;
    f.u[0] = *reinterpret_cast<const ushort8*>(p);
    f.u[1] = *reinterpret_cast<const ushort8*>(p + 8);
    return f.v;
}

// ---------------------------------------------------------------------------
// Prep kernels
// ---------------------------------------------------------------------------
// Pack f32 weight (d x d, row-major, K x N) into WMMA B-operand lane layout,
// zero-padded to Dp x Dp.  grid=(Dp/32, Dp/16), block=32.
__global__ __launch_bounds__(32) void pack_w_kernel(const float* __restrict__ W,
                                                    unsigned short* __restrict__ Wp,
                                                    int d) {
    int lane = threadIdx.x, kt = blockIdx.x, nt = blockIdx.y;
    int half = lane >> 4, n = nt * 16 + (lane & 15);
    unsigned short* out = Wp + ((size_t)(kt * gridDim.y + nt) * 32 + lane) * 16;
    for (int i = 0; i < 16; ++i) {
        int kk = kt * 32 + kmap(i, half);
        float v = (kk < d && n < d) ? W[(size_t)kk * d + n] : 0.f;
        out[i] = f2bf(v);
    }
}

// Pack P^T (B[k=j][n=i] = P[i][j]) from row-major bf16 P (Dp x Dp).
__global__ __launch_bounds__(32) void pack_pT_kernel(const unsigned short* __restrict__ P,
                                                     unsigned short* __restrict__ Pp,
                                                     int Dp) {
    int lane = threadIdx.x, kt = blockIdx.x, nt = blockIdx.y;
    int half = lane >> 4, n = nt * 16 + (lane & 15);
    unsigned short* out = Pp + ((size_t)(kt * gridDim.y + nt) * 32 + lane) * 16;
    for (int i = 0; i < 16; ++i) {
        int kk = kt * 32 + kmap(i, half);
        out[i] = P[(size_t)n * Dp + kk];
    }
}

// Atlas A-matrix: compose the nearest-resample chain analytically.
// Grids: G0=(5,6,4) atlas, G1=(5,7,4) axi, G2=(4,4,7) cor, G3=(4,8,5) sag.
__global__ void build_A_kernel(const float* __restrict__ atlas,
                               unsigned short* __restrict__ A,
                               int Dp, int d, int br, int wz, int wy, int wx) {
    int idx = blockIdx.x * 256 + threadIdx.x;      // m * Dp + j
    int j = idx % Dp, m = idx / Dp;
    unsigned short val = 0;
    if (j < d) {
        const int GZ[4] = {5, 5, 4, 4}, GY[4] = {6, 7, 4, 8}, GX[4] = {4, 4, 7, 5};
        int z = j / (wy * wx), rem = j % (wy * wx), y = rem / wx, x = rem % wx;
        for (int t = br + 1; t > 0; --t) {         // chain down to atlas grid
            z = z * GZ[t - 1] / GZ[t];
            y = y * GY[t - 1] / GY[t];
            x = x * GX[t - 1] / GX[t];
        }
        int c = m / 1000, tt = m % 1000;
        int uz = tt / 100, uy = (tt / 10) % 10, ux = tt % 10;
        int za = uz * 5 + z - 2, ya = uy * 6 + y - 2, xa = ux * 4 + x - 1;
        float f = 0.f;
        if (za >= 0 && za < 46 && ya >= 0 && ya < 56 && xa >= 0 && xa < 38)
            f = atlas[((size_t)(c * 46 + za) * 56 + ya) * 38 + xa];
        val = f2bf(f);
    }
    A[idx] = val;
}

// Feature B-matrix: pad + unfold + view-permute as a gather.
__global__ void build_B_kernel(const float* __restrict__ feat,
                               unsigned short* __restrict__ B,
                               int Dp, int d, int br, int wz, int wy, int wx,
                               int DZ, int DY, int DX, int pbz, int pby, int pbx) {
    int idx = blockIdx.x * 256 + threadIdx.x;
    int j = idx % Dp, m = idx / Dp;
    unsigned short val = 0;
    if (j < d) {
        int vz = j / (wy * wx), rem = j % (wy * wx), vy = rem / wx, vx = rem % wx;
        int c = m / 1000, tt = m % 1000;
        int uz = tt / 100, uy = (tt / 10) % 10, ux = tt % 10;
        int z1 = uz * wz + vz - pbz;
        int y1 = uy * wy + vy - pby;
        int x1 = ux * wx + vx - pbx;
        float f = 0.f;
        if (z1 >= 0 && z1 < DZ && y1 >= 0 && y1 < DY && x1 >= 0 && x1 < DX)
            f = feat[perm_index(br, c, z1, y1, x1)];
        val = f2bf(f);
    }
    B[idx] = val;
}

__global__ void zero_f32_kernel(float* __restrict__ p, int n) {
    int i = blockIdx.x * 256 + threadIdx.x;
    if (i < n) p[i] = 0.f;
}

// ---------------------------------------------------------------------------
// GEMM 1: Out(M x DP) = In(M x DP) * Wp(DP x DP, packed) + bias   (bf16 out)
// Each wave owns a 16 x (NT*16) strip: the A-fragment is loaded ONCE per
// K-step and reused for NT WMMAs (NT = DP/32 accumulators, ~40 VGPRs).
// grid=(M/16, 2), block=32.
// ---------------------------------------------------------------------------
template <int DP>
__global__ __launch_bounds__(32) void gemm_qkv_kernel(const unsigned short* __restrict__ In,
                                                      const unsigned short* __restrict__ Wp,
                                                      const float* __restrict__ bias,
                                                      unsigned short* __restrict__ Out,
                                                      int d) {
    constexpr int KT = DP / 32;         // K-steps
    constexpr int NT = DP / 32;         // N-tiles per wave (grid.y = 2 strips)
    int m0 = blockIdx.x * 16;
    int nt0 = blockIdx.y * NT;
    int lane = threadIdx.x;
    v8f acc[NT] = {};
    for (int kt = 0; kt < KT; ++kt) {
        v16bf a = load_frag_rowmajor<DP>(In + (size_t)m0 * DP, kt * 32, lane);
#pragma unroll
        for (int nt = 0; nt < NT; ++nt) {
            v16bf b = load_frag_packed<DP>(Wp, kt, nt0 + nt, lane);
            acc[nt] = __builtin_amdgcn_wmma_f32_16x16x32_bf16(false, a, false, b,
                                                              (short)0, acc[nt], false, false);
        }
    }
    int half = lane >> 4;
#pragma unroll
    for (int nt = 0; nt < NT; ++nt) {
        int n = (nt0 + nt) * 16 + (lane & 15);
        float bi = (n < d) ? bias[n] : 0.f;
#pragma unroll
        for (int r = 0; r < 8; ++r) {
            int m = m0 + r + 8 * half;
            Out[(size_t)m * DP + n] = f2bf(acc[nt][r] + bi);
        }
    }
}

// ---------------------------------------------------------------------------
// GEMM 2 (scores): S(Dp x Dp) += q^T * k, reduction over M (split-M atomics).
// Coalesced B128 stage into LDS, WMMA-layout gather from LDS.
// grid=(Dp/16, Dp/16, SPLIT), block=32.
// ---------------------------------------------------------------------------
__global__ __launch_bounds__(32) void gemm_scores_kernel(const unsigned short* __restrict__ Q,
                                                         const unsigned short* __restrict__ K,
                                                         float* __restrict__ S,
                                                         int Dp, int mChunk) {
    __shared__ unsigned short qs[32 * 16];
    __shared__ unsigned short ks[32 * 16];
    int qt = blockIdx.x, ktile = blockIdx.y, lane = threadIdx.x;
    int mBeg = blockIdx.z * mChunk;
    int half = lane >> 4, nr = lane & 15;
    v8f acc = {};
    for (int mb = mBeg; mb < mBeg + mChunk; mb += 32) {
        const unsigned short* qp = Q + (size_t)(mb + lane) * Dp + qt * 16;
        const unsigned short* kp = K + (size_t)(mb + lane) * Dp + ktile * 16;
        *reinterpret_cast<ushort8*>(&qs[lane * 16])     = *reinterpret_cast<const ushort8*>(qp);
        *reinterpret_cast<ushort8*>(&qs[lane * 16 + 8]) = *reinterpret_cast<const ushort8*>(qp + 8);
        *reinterpret_cast<ushort8*>(&ks[lane * 16])     = *reinterpret_cast<const ushort8*>(kp);
        *reinterpret_cast<ushort8*>(&ks[lane * 16 + 8]) = *reinterpret_cast<const ushort8*>(kp + 8);
        __syncthreads();
        FragBF fa, fb;
#pragma unroll
        for (int i = 0; i < 16; ++i) {
            int km = kmap(i, half);              // m-offset within the 32-slab
            fa.s[i] = qs[km * 16 + nr];          // A row = q-feature, K = m
            fb.s[i] = ks[km * 16 + nr];          // B col = k-feature, K = m
        }
        acc = __builtin_amdgcn_wmma_f32_16x16x32_bf16(false, fa.v, false, fb.v,
                                                      (short)0, acc, false, false);
        __syncthreads();
    }
    int i0 = qt * 16, j0 = ktile * 16;
#pragma unroll
    for (int r = 0; r < 8; ++r)
        atomicAdd(&S[(size_t)(i0 + r + 8 * half) * Dp + (j0 + nr)], acc[r]);
}

// Row softmax over valid d x d region; bf16 output, zero in padding.
__global__ __launch_bounds__(32) void softmax_kernel(const float* __restrict__ S,
                                                     unsigned short* __restrict__ P,
                                                     int Dp, int d) {
    int i = blockIdx.x, lane = threadIdx.x;
    if (i >= d) {
        for (int j = lane; j < Dp; j += 32) P[(size_t)i * Dp + j] = 0;
        return;
    }
    float mx = -3.4e38f;
    for (int j = lane; j < d; j += 32) mx = fmaxf(mx, S[(size_t)i * Dp + j]);
    for (int o = 16; o > 0; o >>= 1)   mx = fmaxf(mx, __shfl_xor(mx, o, 32));
    float sum = 0.f;
    for (int j = lane; j < d; j += 32) sum += __expf(S[(size_t)i * Dp + j] - mx);
    for (int o = 16; o > 0; o >>= 1)   sum += __shfl_xor(sum, o, 32);
    float inv = 1.f / sum;
    for (int j = lane; j < Dp; j += 32) {
        float pv = (j < d) ? __expf(S[(size_t)i * Dp + j] - mx) * inv : 0.f;
        P[(size_t)i * Dp + j] = f2bf(pv);
    }
}

// ---------------------------------------------------------------------------
// GEMM 3 (cross) fused with fold + crop + inverse-permute epilogue.
// cross[m,i] = sum_j v[m,j] * P[i,j]  ->  out[...] (f32).
// Same NT-strip structure as gemm_qkv.  grid=(M/16, 2), block=32.
// ---------------------------------------------------------------------------
template <int DP>
__global__ __launch_bounds__(32) void gemm_cross_fold_kernel(const unsigned short* __restrict__ V,
                                                             const unsigned short* __restrict__ Pp,
                                                             float* __restrict__ out,
                                                             int d, int br,
                                                             int wz, int wy, int wx,
                                                             int DZ, int DY, int DX,
                                                             int pbz, int pby, int pbx) {
    constexpr int KT = DP / 32;
    constexpr int NT = DP / 32;
    int m0 = blockIdx.x * 16;
    int nt0 = blockIdx.y * NT;
    int lane = threadIdx.x;
    v8f acc[NT] = {};
    for (int kt = 0; kt < KT; ++kt) {
        v16bf a = load_frag_rowmajor<DP>(V + (size_t)m0 * DP, kt * 32, lane);
#pragma unroll
        for (int nt = 0; nt < NT; ++nt) {
            v16bf b = load_frag_packed<DP>(Pp, kt, nt0 + nt, lane);
            acc[nt] = __builtin_amdgcn_wmma_f32_16x16x32_bf16(false, a, false, b,
                                                              (short)0, acc[nt], false, false);
        }
    }
    int half = lane >> 4;
#pragma unroll
    for (int nt = 0; nt < NT; ++nt) {
        int i = (nt0 + nt) * 16 + (lane & 15);
        if (i >= d) continue;
        int vz = i / (wy * wx), rem = i % (wy * wx), vy = rem / wx, vx = rem % wx;
#pragma unroll
        for (int r = 0; r < 8; ++r) {
            int m = m0 + r + 8 * half;
            int c = m / 1000, tt = m % 1000;
            int uz = tt / 100, uy = (tt / 10) % 10, ux = tt % 10;
            int z1 = uz * wz + vz - pbz;
            int y1 = uy * wy + vy - pby;
            int x1 = ux * wx + vx - pbx;
            if (z1 >= 0 && z1 < DZ && y1 >= 0 && y1 < DY && x1 >= 0 && x1 < DX)
                out[perm_index(br, c, z1, y1, x1)] = acc[nt][r];
        }
    }
}

// ---------------------------------------------------------------------------
// Host launcher
// ---------------------------------------------------------------------------
extern "C" void kernel_launch(void* const* d_in, const int* in_sizes, int n_in,
                              void* d_out, int out_size, void* d_ws, size_t ws_size,
                              hipStream_t stream) {
    (void)in_sizes; (void)n_in; (void)out_size; (void)ws_size;

    const int M = 64000;
    const size_t CAP = (size_t)M * 160;            // ushorts per big buffer
    unsigned short* ws = (unsigned short*)d_ws;
    unsigned short* Abuf = ws;
    unsigned short* Bbuf = Abuf + CAP;
    unsigned short* Qbuf = Bbuf + CAP;
    unsigned short* Kbuf = Qbuf + CAP;
    unsigned short* Vbuf = Abuf;                   // V overwrites A after q-GEMM
    unsigned short* Wqp  = Kbuf + CAP;
    unsigned short* Wkp  = Wqp + 160 * 160;
    unsigned short* Wvp  = Wkp + 160 * 160;
    float*          S    = (float*)(Wvp + 160 * 160);
    unsigned short* Prm  = (unsigned short*)(S + 160 * 160);
    unsigned short* Ppk  = Prm + 160 * 160;

    const float* atlas = (const float*)d_in[3];
    float*       out   = (float*)d_out;

    struct Br {
        const float* feat; int widx; size_t outOff;
        int wz, wy, wx, d, Dp;
        int DZ, DY, DX, pbz, pby, pbx;
    };
    const Br brs[3] = {
        { (const float*)d_in[0],  4, 0,               5, 7, 4, 140, 160, 48, 66, 38, 1, 2, 1 },
        { (const float*)d_in[1], 10, (size_t)7704576,  4, 4, 7, 112, 128, 38, 38, 66, 1, 1, 2 },
        { (const float*)d_in[2], 16, (size_t)13804032, 4, 8, 5, 160, 160, 38, 78, 48, 1, 1, 1 },
    };

    for (int br = 0; br < 3; ++br) {
        const Br& b = brs[br];
        const int Dp = b.Dp, d = b.d;
        const int ktiles = Dp / 32, ntiles = Dp / 16;
        const float* Wq = (const float*)d_in[b.widx + 0];
        const float* bq = (const float*)d_in[b.widx + 1];
        const float* Wk = (const float*)d_in[b.widx + 2];
        const float* bk = (const float*)d_in[b.widx + 3];
        const float* Wv = (const float*)d_in[b.widx + 4];
        const float* bv = (const float*)d_in[b.widx + 5];

        dim3 pg(ktiles, ntiles);
        pack_w_kernel<<<pg, 32, 0, stream>>>(Wq, Wqp, d);
        pack_w_kernel<<<pg, 32, 0, stream>>>(Wk, Wkp, d);
        pack_w_kernel<<<pg, 32, 0, stream>>>(Wv, Wvp, d);

        int nElem = M * Dp;
        build_A_kernel<<<nElem / 256, 256, 0, stream>>>(atlas, Abuf, Dp, d, br,
                                                        b.wz, b.wy, b.wx);
        build_B_kernel<<<nElem / 256, 256, 0, stream>>>(b.feat, Bbuf, Dp, d, br,
                                                        b.wz, b.wy, b.wx,
                                                        b.DZ, b.DY, b.DX,
                                                        b.pbz, b.pby, b.pbx);

        dim3 gg(M / 16, 2);
        if (Dp == 160) {
            gemm_qkv_kernel<160><<<gg, 32, 0, stream>>>(Abuf, Wqp, bq, Qbuf, d);
            gemm_qkv_kernel<160><<<gg, 32, 0, stream>>>(Bbuf, Wkp, bk, Kbuf, d);
            gemm_qkv_kernel<160><<<gg, 32, 0, stream>>>(Bbuf, Wvp, bv, Vbuf, d);
        } else {
            gemm_qkv_kernel<128><<<gg, 32, 0, stream>>>(Abuf, Wqp, bq, Qbuf, d);
            gemm_qkv_kernel<128><<<gg, 32, 0, stream>>>(Bbuf, Wkp, bk, Kbuf, d);
            gemm_qkv_kernel<128><<<gg, 32, 0, stream>>>(Bbuf, Wvp, bv, Vbuf, d);
        }

        zero_f32_kernel<<<(Dp * Dp + 255) / 256, 256, 0, stream>>>(S, Dp * Dp);
        const int SPLIT = 125, mChunk = M / SPLIT;            // 512 rows, 16 WMMA steps
        dim3 sg(ntiles, ntiles, SPLIT);
        gemm_scores_kernel<<<sg, 32, 0, stream>>>(Qbuf, Kbuf, S, Dp, mChunk);

        softmax_kernel<<<Dp, 32, 0, stream>>>(S, Prm, Dp, d);
        pack_pT_kernel<<<pg, 32, 0, stream>>>(Prm, Ppk, Dp);

        if (Dp == 160) {
            gemm_cross_fold_kernel<160><<<gg, 32, 0, stream>>>(Vbuf, Ppk, out + b.outOff,
                                                               d, br,
                                                               b.wz, b.wy, b.wx,
                                                               b.DZ, b.DY, b.DX,
                                                               b.pbz, b.pby, b.pbx);
        } else {
            gemm_cross_fold_kernel<128><<<gg, 32, 0, stream>>>(Vbuf, Ppk, out + b.outOff,
                                                               d, br,
                                                               b.wz, b.wy, b.wx,
                                                               b.DZ, b.DY, b.DX,
                                                               b.pbz, b.pby, b.pbx);
        }
    }
}
